// ScannedLRU_87591563035254
// MI455X (gfx1250) — matrix operand order repaired and checked by reading, not compiled, not run on hardware
//
#include <hip/hip_runtime.h>

#define T_LEN   2048
#define BATCH   16
#define DM      512
#define HDIM    512
#define M_TOT   (T_LEN * BATCH)     // 32768
#define NCHUNK  16
#define CHUNK   128                 // T per chunk

typedef __bf16 bf16x16 __attribute__((ext_vector_type(16)));
typedef float  f32x8   __attribute__((ext_vector_type(8)));

union ABTile { bf16x16 v; uint4 q[2]; };

__device__ __forceinline__ unsigned short bf16_rne(float f) {
    unsigned u = __float_as_uint(f);
    unsigned r = u + 0x7FFFu + ((u >> 16) & 1u);
    return (unsigned short)(r >> 16);
}
__device__ __forceinline__ float bf16_f32(unsigned short h) {
    return __uint_as_float(((unsigned)h) << 16);
}
__device__ __forceinline__ void split2(float v, unsigned short* hi, unsigned short* lo) {
    unsigned short h = bf16_rne(v);
    *hi = h;
    *lo = bf16_rne(v - bf16_f32(h));
}

// ---------------- prep: Lambda, gamma-scaled B, C (with -C_im), split to bf16 hi/lo ---
__global__ void __launch_bounds__(256) prep_weights(
    const float* __restrict__ nu_log, const float* __restrict__ theta_log,
    const float* __restrict__ B_re,   const float* __restrict__ B_im,
    const float* __restrict__ C_re,   const float* __restrict__ C_im,
    float* __restrict__ lam_re, float* __restrict__ lam_im,
    unsigned short* __restrict__ wbre_hi, unsigned short* __restrict__ wbre_lo,
    unsigned short* __restrict__ wbim_hi, unsigned short* __restrict__ wbim_lo,
    unsigned short* __restrict__ wcre_hi, unsigned short* __restrict__ wcre_lo,
    unsigned short* __restrict__ wcin_hi, unsigned short* __restrict__ wcin_lo)
{
    int i = blockIdx.x * blockDim.x + threadIdx.x;   // 0 .. H*D-1
    if (i < HDIM) {
        float nu  = __expf(nu_log[i]);
        float th  = __expf(theta_log[i]);
        float mag = __expf(-nu);
        lam_re[i] = mag * __cosf(th);
        lam_im[i] = mag * __sinf(th);
    }
    int h = i >> 9;                                  // row of (H, D) B matrices
    float nu_h  = __expf(nu_log[h]);
    float gamma = sqrtf(1.0f - __expf(-2.0f * nu_h));
    split2(B_re[i] * gamma, &wbre_hi[i], &wbre_lo[i]);
    split2(B_im[i] * gamma, &wbim_hi[i], &wbim_lo[i]);
    split2(C_re[i],  &wcre_hi[i], &wcre_lo[i]);
    split2(-C_im[i], &wcin_hi[i], &wcin_lo[i]);      // pre-negated imag part of C
}

// ---------------- prep: split x into bf16 hi/lo ----------------
__global__ void __launch_bounds__(256) prep_x(
    const float* __restrict__ x,
    unsigned short* __restrict__ x_hi, unsigned short* __restrict__ x_lo)
{
    int gid = blockIdx.x * blockDim.x + threadIdx.x;
    int base = gid * 4;
    float4 v = *(const float4*)(x + base);
    ushort4 hi, lo;
    split2(v.x, &hi.x, &lo.x);
    split2(v.y, &hi.y, &lo.y);
    split2(v.z, &hi.z, &lo.z);
    split2(v.w, &hi.w, &lo.w);
    *(ushort4*)(x_hi + base) = hi;
    *(ushort4*)(x_lo + base) = lo;
}

#define WMMA_BF16(A, B, C) \
    __builtin_amdgcn_wmma_f32_16x16x32_bf16(false, (A), false, (B), (short)0, (C), false, false)

// ---------------- GEMM 1: Bu = x @ (gamma*B)^T  (re & im), split-bf16 x3 WMMA -------
__global__ void __launch_bounds__(256) gemm_bu(
    const unsigned short* __restrict__ x_hi, const unsigned short* __restrict__ x_lo,
    const unsigned short* __restrict__ wre_hi, const unsigned short* __restrict__ wre_lo,
    const unsigned short* __restrict__ wim_hi, const unsigned short* __restrict__ wim_lo,
    float* __restrict__ bu_re, float* __restrict__ bu_im)
{
    const int lane  = threadIdx.x & 31;
    const int gwave = (blockIdx.x * blockDim.x + threadIdx.x) >> 5;
    const int mt = gwave >> 3;          // 0..2047 : M tile
    const int ng = gwave & 7;           // 0..7    : group of 4 N tiles
    const int m0 = mt << 4;
    const int n0 = ng << 6;

    // A operand: lane m = lane&15, contiguous K chunk [kb16, kb16+16), same k order as B
    const int am  = m0 + (lane & 15);
    const int akb = (lane >> 4) << 4;
    const unsigned short* aph = x_hi + am * DM + akb;
    const unsigned short* apl = x_lo + am * DM + akb;

    // B operand: lane n = lane&15, contiguous K chunk [kb16, kb16+16), same k order as A
    const int bn   = lane & 15;
    const int bkb  = (lane >> 4) << 4;

    f32x8 acc_re[4], acc_im[4];
    #pragma unroll
    for (int j = 0; j < 4; ++j) { acc_re[j] = (f32x8)(0.0f); acc_im[j] = (f32x8)(0.0f); }

    #pragma unroll 2
    for (int k = 0; k < DM; k += 32) {
        ABTile ah, al;
        ah.q[0] = *(const uint4*)(aph + k);
        ah.q[1] = *(const uint4*)(aph + k + 8);
        al.q[0] = *(const uint4*)(apl + k);
        al.q[1] = *(const uint4*)(apl + k + 8);
        #pragma unroll
        for (int j = 0; j < 4; ++j) {
            const int brow = (n0 + (j << 4) + bn) * DM + k + bkb;
            ABTile brh, brl, bih, bil;
            brh.q[0] = *(const uint4*)(wre_hi + brow);
            brh.q[1] = *(const uint4*)(wre_hi + brow + 8);
            brl.q[0] = *(const uint4*)(wre_lo + brow);
            brl.q[1] = *(const uint4*)(wre_lo + brow + 8);
            bih.q[0] = *(const uint4*)(wim_hi + brow);
            bih.q[1] = *(const uint4*)(wim_hi + brow + 8);
            bil.q[0] = *(const uint4*)(wim_lo + brow);
            bil.q[1] = *(const uint4*)(wim_lo + brow + 8);
            acc_re[j] = WMMA_BF16(ah.v, brh.v, acc_re[j]);
            acc_re[j] = WMMA_BF16(ah.v, brl.v, acc_re[j]);
            acc_re[j] = WMMA_BF16(al.v, brh.v, acc_re[j]);
            acc_im[j] = WMMA_BF16(ah.v, bih.v, acc_im[j]);
            acc_im[j] = WMMA_BF16(ah.v, bil.v, acc_im[j]);
            acc_im[j] = WMMA_BF16(al.v, bih.v, acc_im[j]);
        }
    }

    // C/D layout: lane n = lane&15; VGPR r -> m = r (+8 for upper lanes)
    const int cn = lane & 15;
    const int rb = (lane >> 4) << 3;
    #pragma unroll
    for (int j = 0; j < 4; ++j) {
        #pragma unroll
        for (int r = 0; r < 8; ++r) {
            int m = m0 + rb + r;
            int col = n0 + (j << 4) + cn;
            bu_re[m * HDIM + col] = acc_re[j][r];
            bu_im[m * HDIM + col] = acc_im[j][r];
        }
    }
}

// ---------------- scan pass A: per-chunk local final state (h0 = 0) ----------------
__global__ void __launch_bounds__(256) scan_chunk(
    const float* __restrict__ bu_re, const float* __restrict__ bu_im,
    const float* __restrict__ lam_re, const float* __restrict__ lam_im,
    float* __restrict__ cf_re, float* __restrict__ cf_im)
{
    int gid = blockIdx.x * blockDim.x + threadIdx.x;   // c*8192 + b*512 + h
    int h = gid & (HDIM - 1);
    int b = (gid >> 9) & (BATCH - 1);
    int c = gid >> 13;
    float lr = lam_re[h], li = lam_im[h];
    float sr = 0.0f, si = 0.0f;
    int idx = ((c * CHUNK) * BATCH + b) * HDIM + h;
    const int stride = BATCH * HDIM;
    for (int tl = 0; tl < CHUNK; ++tl) {
        float br = bu_re[idx], bi = bu_im[idx];
        float nr = fmaf(lr, sr, fmaf(-li, si, br));
        float ni = fmaf(lr, si, fmaf(li, sr, bi));
        sr = nr; si = ni;
        idx += stride;
    }
    cf_re[gid] = sr; cf_im[gid] = si;
}

// ---------------- scan pass B: chunk-level carry scan with Lam^128; final state out --
__global__ void __launch_bounds__(256) scan_carry(
    const float* __restrict__ cf_re, const float* __restrict__ cf_im,
    const float* __restrict__ lam_re, const float* __restrict__ lam_im,
    const float* __restrict__ h0_re, const float* __restrict__ h0_im,
    float* __restrict__ carry_re, float* __restrict__ carry_im,
    float* __restrict__ final_out)
{
    int gid = blockIdx.x * blockDim.x + threadIdx.x;   // b*512 + h
    int h = gid & (HDIM - 1);
    float pr = lam_re[h], pi = lam_im[h];
    #pragma unroll
    for (int s = 0; s < 7; ++s) {                      // Lam^128 by squaring
        float nr = pr * pr - pi * pi;
        float ni = 2.0f * pr * pi;
        pr = nr; pi = ni;
    }
    float sr = h0_re[gid], si = h0_im[gid];
    #pragma unroll
    for (int c = 0; c < NCHUNK; ++c) {
        carry_re[c * 8192 + gid] = sr;
        carry_im[c * 8192 + gid] = si;
        float fr = cf_re[c * 8192 + gid], fi = cf_im[c * 8192 + gid];
        float nr = fmaf(pr, sr, fmaf(-pi, si, fr));
        float ni = fmaf(pr, si, fmaf(pi, sr, fi));
        sr = nr; si = ni;
    }
    final_out[gid * 2]     = sr;                        // complex64 interleaved
    final_out[gid * 2 + 1] = si;
}

// ---------------- scan pass C: replay chunk with true carry, emit split-bf16 h ------
__global__ void __launch_bounds__(256) scan_apply(
    const float* __restrict__ bu_re, const float* __restrict__ bu_im,
    const float* __restrict__ lam_re, const float* __restrict__ lam_im,
    const float* __restrict__ carry_re, const float* __restrict__ carry_im,
    unsigned short* __restrict__ hre_hi, unsigned short* __restrict__ hre_lo,
    unsigned short* __restrict__ him_hi, unsigned short* __restrict__ him_lo)
{
    int gid = blockIdx.x * blockDim.x + threadIdx.x;
    int h = gid & (HDIM - 1);
    int b = (gid >> 9) & (BATCH - 1);
    int c = gid >> 13;
    float lr = lam_re[h], li = lam_im[h];
    float sr = carry_re[gid], si = carry_im[gid];
    int idx = ((c * CHUNK) * BATCH + b) * HDIM + h;
    const int stride = BATCH * HDIM;
    for (int tl = 0; tl < CHUNK; ++tl) {
        float br = bu_re[idx], bi = bu_im[idx];
        float nr = fmaf(lr, sr, fmaf(-li, si, br));
        float ni = fmaf(lr, si, fmaf(li, sr, bi));
        sr = nr; si = ni;
        split2(sr, &hre_hi[idx], &hre_lo[idx]);
        split2(si, &him_hi[idx], &him_lo[idx]);
        idx += stride;
    }
}

// ---------------- GEMM 2: y = h_re@C_re^T + h_im@(-C_im)^T + D*x --------------------
__global__ void __launch_bounds__(256) gemm_y(
    const unsigned short* __restrict__ hre_hi, const unsigned short* __restrict__ hre_lo,
    const unsigned short* __restrict__ him_hi, const unsigned short* __restrict__ him_lo,
    const unsigned short* __restrict__ wcre_hi, const unsigned short* __restrict__ wcre_lo,
    const unsigned short* __restrict__ wcin_hi, const unsigned short* __restrict__ wcin_lo,
    const float* __restrict__ x, const float* __restrict__ D_vec,
    float* __restrict__ y_out)
{
    const int lane  = threadIdx.x & 31;
    const int gwave = (blockIdx.x * blockDim.x + threadIdx.x) >> 5;
    const int mt = gwave >> 3;
    const int ng = gwave & 7;
    const int m0 = mt << 4;
    const int n0 = ng << 6;

    const int am  = m0 + (lane & 15);
    const int akb = (lane >> 4) << 4;
    const unsigned short* prh = hre_hi + am * HDIM + akb;
    const unsigned short* prl = hre_lo + am * HDIM + akb;
    const unsigned short* pih = him_hi + am * HDIM + akb;
    const unsigned short* pil = him_lo + am * HDIM + akb;

    const int bn  = lane & 15;
    const int bkb = (lane >> 4) << 4;

    f32x8 acc[4];
    #pragma unroll
    for (int j = 0; j < 4; ++j) acc[j] = (f32x8)(0.0f);

    #pragma unroll 2
    for (int k = 0; k < HDIM; k += 32) {
        ABTile arh, arl, aih, ail;
        arh.q[0] = *(const uint4*)(prh + k); arh.q[1] = *(const uint4*)(prh + k + 8);
        arl.q[0] = *(const uint4*)(prl + k); arl.q[1] = *(const uint4*)(prl + k + 8);
        aih.q[0] = *(const uint4*)(pih + k); aih.q[1] = *(const uint4*)(pih + k + 8);
        ail.q[0] = *(const uint4*)(pil + k); ail.q[1] = *(const uint4*)(pil + k + 8);
        #pragma unroll
        for (int j = 0; j < 4; ++j) {
            const int brow = (n0 + (j << 4) + bn) * HDIM + k + bkb;
            ABTile crh, crl, cih, cil;
            crh.q[0] = *(const uint4*)(wcre_hi + brow);
            crh.q[1] = *(const uint4*)(wcre_hi + brow + 8);
            crl.q[0] = *(const uint4*)(wcre_lo + brow);
            crl.q[1] = *(const uint4*)(wcre_lo + brow + 8);
            cih.q[0] = *(const uint4*)(wcin_hi + brow);
            cih.q[1] = *(const uint4*)(wcin_hi + brow + 8);
            cil.q[0] = *(const uint4*)(wcin_lo + brow);
            cil.q[1] = *(const uint4*)(wcin_lo + brow + 8);
            acc[j] = WMMA_BF16(arh.v, crh.v, acc[j]);
            acc[j] = WMMA_BF16(arh.v, crl.v, acc[j]);
            acc[j] = WMMA_BF16(arl.v, crh.v, acc[j]);
            acc[j] = WMMA_BF16(aih.v, cih.v, acc[j]);
            acc[j] = WMMA_BF16(aih.v, cil.v, acc[j]);
            acc[j] = WMMA_BF16(ail.v, cih.v, acc[j]);
        }
    }

    const int cn = lane & 15;
    const int rb = (lane >> 4) << 3;
    #pragma unroll
    for (int j = 0; j < 4; ++j) {
        int col = n0 + (j << 4) + cn;
        float dv = D_vec[col];
        #pragma unroll
        for (int r = 0; r < 8; ++r) {
            int m = m0 + rb + r;
            y_out[m * DM + col] = acc[j][r] + dv * x[m * DM + col];
        }
    }
}

extern "C" void kernel_launch(void* const* d_in, const int* in_sizes, int n_in,
                              void* d_out, int out_size, void* d_ws, size_t ws_size,
                              hipStream_t stream) {
    (void)in_sizes; (void)n_in; (void)out_size; (void)ws_size;
    const float* x         = (const float*)d_in[0];
    const float* h0_re     = (const float*)d_in[1];
    const float* h0_im     = (const float*)d_in[2];
    const float* nu_log    = (const float*)d_in[3];
    const float* theta_log = (const float*)d_in[4];
    const float* B_re      = (const float*)d_in[5];
    const float* B_im      = (const float*)d_in[6];
    const float* C_re      = (const float*)d_in[7];
    const float* C_im      = (const float*)d_in[8];
    const float* D_vec     = (const float*)d_in[9];

    char* ws = (char*)d_ws;
    size_t off = 0;
    auto grab = [&](size_t bytes) -> char* {
        char* p = ws + off;
        off = (off + bytes + 255) & ~(size_t)255;
        return p;
    };
    const size_t W_ELEMS = (size_t)HDIM * DM;          // 262144
    const size_t M_ELEMS = (size_t)M_TOT * HDIM;       // 16.7M

    float* lam_re = (float*)grab(HDIM * 4);
    float* lam_im = (float*)grab(HDIM * 4);
    unsigned short* wbre_hi = (unsigned short*)grab(W_ELEMS * 2);
    unsigned short* wbre_lo = (unsigned short*)grab(W_ELEMS * 2);
    unsigned short* wbim_hi = (unsigned short*)grab(W_ELEMS * 2);
    unsigned short* wbim_lo = (unsigned short*)grab(W_ELEMS * 2);
    unsigned short* wcre_hi = (unsigned short*)grab(W_ELEMS * 2);
    unsigned short* wcre_lo = (unsigned short*)grab(W_ELEMS * 2);
    unsigned short* wcin_hi = (unsigned short*)grab(W_ELEMS * 2);
    unsigned short* wcin_lo = (unsigned short*)grab(W_ELEMS * 2);
    unsigned short* x_hi    = (unsigned short*)grab(M_ELEMS * 2);
    unsigned short* x_lo    = (unsigned short*)grab(M_ELEMS * 2);
    float* bu_re = (float*)grab(M_ELEMS * 4);
    float* bu_im = (float*)grab(M_ELEMS * 4);
    unsigned short* him_hi = (unsigned short*)grab(M_ELEMS * 2);
    unsigned short* him_lo = (unsigned short*)grab(M_ELEMS * 2);
    float* cf_re    = (float*)grab(NCHUNK * 8192 * 4);
    float* cf_im    = (float*)grab(NCHUNK * 8192 * 4);
    float* carry_re = (float*)grab(NCHUNK * 8192 * 4);
    float* carry_im = (float*)grab(NCHUNK * 8192 * 4);
    // h_re (split) reuses x buffers: x is no longer needed in bf16 form after GEMM 1.
    unsigned short* hre_hi = x_hi;
    unsigned short* hre_lo = x_lo;

    float* final_out = (float*)d_out;                  // 8192 complex64 -> 16384 floats
    float* y_out     = (float*)d_out + 2 * BATCH * HDIM;

    prep_weights<<<W_ELEMS / 256, 256, 0, stream>>>(
        nu_log, theta_log, B_re, B_im, C_re, C_im,
        lam_re, lam_im, wbre_hi, wbre_lo, wbim_hi, wbim_lo,
        wcre_hi, wcre_lo, wcin_hi, wcin_lo);

    prep_x<<<M_ELEMS / (256 * 4), 256, 0, stream>>>(x, x_hi, x_lo);

    gemm_bu<<<2048, 256, 0, stream>>>(x_hi, x_lo, wbre_hi, wbre_lo, wbim_hi, wbim_lo,
                                      bu_re, bu_im);

    scan_chunk<<<(NCHUNK * BATCH * HDIM) / 256, 256, 0, stream>>>(
        bu_re, bu_im, lam_re, lam_im, cf_re, cf_im);

    scan_carry<<<(BATCH * HDIM) / 256, 256, 0, stream>>>(
        cf_re, cf_im, lam_re, lam_im, h0_re, h0_im, carry_re, carry_im, final_out);

    scan_apply<<<(NCHUNK * BATCH * HDIM) / 256, 256, 0, stream>>>(
        bu_re, bu_im, lam_re, lam_im, carry_re, carry_im,
        hre_hi, hre_lo, him_hi, him_lo);

    gemm_y<<<2048, 256, 0, stream>>>(hre_hi, hre_lo, him_hi, him_lo,
                                     wcre_hi, wcre_lo, wcin_hi, wcin_lo,
                                     x, D_vec, y_out);
}
